// Encoder_37417755082985
// MI455X (gfx1250) — matrix operand order repaired and checked by reading, not compiled, not run on hardware
//
#include <hip/hip_runtime.h>
#include <hip/hip_bf16.h>

// ---------------- Problem constants ----------------
#define PAD   0
#define BB    32
#define SS    512
#define DD    256
#define HH    512
#define NHH   8
#define HDD   64
#define PFF   2048
#define NMAXX 15
#define LL    2
#define NTOK  (BB * SS)          // 16384 rows
#define LSTR  40                 // LDS row stride (ushorts): 80B, 16B-aligned rows

// ---------------- Vector / WMMA types ----------------
typedef __attribute__((ext_vector_type(16))) __bf16 bf16x16;
typedef __attribute__((ext_vector_type(2)))  __bf16 vbf2;
typedef __attribute__((ext_vector_type(8)))  float  f32x8;
typedef __attribute__((ext_vector_type(4)))  float  vf4;
typedef __attribute__((ext_vector_type(2)))  float  vf2;
typedef __attribute__((ext_vector_type(4)))  unsigned vu4;
typedef __attribute__((ext_vector_type(2)))  unsigned vu2;

union FragU { unsigned u[8]; vu4 q[2]; bf16x16 v; };

// native f32 -> bf16 (RNE) hardware converts
__device__ __forceinline__ unsigned short f2bf(float f) {
    union { __bf16 h; unsigned short s; } c;
    c.h = (__bf16)f;
    return c.s;
}
// packed 2xf32 -> 2xbf16 (v_cvt_pk_bf16_f32 when available)
__device__ __forceinline__ unsigned pack2(float lo, float hi) {
    vf2 f; f.x = lo; f.y = hi;
    union { vbf2 v; unsigned u; } c;
    c.v = __builtin_convertvector(f, vbf2);
    return c.u;
}

// ---------------- Embedding + PE + time encoding ----------------
__global__ __launch_bounds__(DD) void embed_kernel(
    const int* __restrict__ src, const float* __restrict__ tint,
    const float* __restrict__ emb, const float* __restrict__ timeW,
    const float* __restrict__ timeB, float* __restrict__ x)
{
    int n = blockIdx.x, d = threadIdx.x;
    int tok = src[n];
    int s = n & (SS - 1);
    int i = d >> 1;
    float div = __expf((float)(2 * i) * (-9.210340371976184f / (float)DD));
    float ang = (float)s * div;
    float pe = (d & 1) ? __cosf(ang) : __sinf(ang);
    float val = emb[(size_t)tok * DD + d] + pe + tint[n] * timeW[d] + timeB[d];
    x[(size_t)n * DD + d] = val;
}

// ---------------- Neighbor aggregation (gather-mean) ----------------
__global__ __launch_bounds__(DD) void agg_kernel(
    const float* __restrict__ x, const int* __restrict__ nidx,
    const int* __restrict__ ncnt, float* __restrict__ agg)
{
    int n = blockIdx.x, d = threadIdx.x;
    int cnt = ncnt[n];
    float s = 0.f;
    for (int j = 0; j < cnt; ++j) {
        int id = nidx[n * NMAXX + j];
        s += x[(size_t)id * DD + d];
    }
    int denom = cnt > 0 ? cnt : 1;
    agg[(size_t)n * DD + d] = s / (float)denom;
}

// ---------------- Generic bf16-WMMA GEMM: C = act(A@B + bias [+ res]) ----------------
// A: [M,K] (f32 or bf16), B: [K,N] f32 weights, out: [M,N] f32 or bf16
// block tile 128x64, BK=32, 8 waves x (16x64)
template <bool A_BF16, bool RELU, bool OUT_BF16, bool HAS_RES>
__global__ __launch_bounds__(256) void gemm_kernel(
    const void* __restrict__ Aptr, const float* __restrict__ Bw,
    const float* __restrict__ bias, const float* __restrict__ res,
    void* __restrict__ Out, int M, int N, int K)
{
    __shared__ unsigned short As[128][LSTR];
    __shared__ unsigned short Bs[64][LSTR];

    int t = threadIdx.x;
    int wave = t >> 5, lane = t & 31;
    int hl = lane >> 4, li = lane & 15;
    int bm = blockIdx.y * 128, bn = blockIdx.x * 64;

    f32x8 acc[4];
    for (int nt = 0; nt < 4; ++nt)
        for (int r = 0; r < 8; ++r) acc[nt][r] = 0.f;

    for (int k0 = 0; k0 < K; k0 += 32) {
        // ---- stage A (128x32) : b128/b64 loads, packed b32 LDS stores ----
        if (A_BF16) {
            const unsigned short* Ab = (const unsigned short*)Aptr;
            for (int i = 0; i < 4; ++i) {
                int qd = i * 256 + t;              // 1024 quads
                int m = qd >> 3, kk = (qd & 7) * 4;
                vu2 d = *(const vu2*)(Ab + (size_t)(bm + m) * K + k0 + kk);
                *(unsigned*)&As[m][kk]     = d.x;
                *(unsigned*)&As[m][kk + 2] = d.y;
            }
        } else {
            const float* Af = (const float*)Aptr;
            for (int i = 0; i < 4; ++i) {
                int qd = i * 256 + t;
                int m = qd >> 3, kk = (qd & 7) * 4;
                vf4 f = *(const vf4*)(Af + (size_t)(bm + m) * K + k0 + kk);
                *(unsigned*)&As[m][kk]     = pack2(f.x, f.y);
                *(unsigned*)&As[m][kk + 2] = pack2(f.z, f.w);
            }
        }
        // ---- stage B transposed (Bs[n][k]) : vertical packs, 1 group/thread ----
        {
            int kp = t >> 4, nq = t & 15;          // 256 groups
            int kk = kp * 2, n = nq * 4;
            const float* r0 = Bw + (size_t)(k0 + kk) * N + bn + n;
            vf4 f0 = *(const vf4*)r0;
            vf4 f1 = *(const vf4*)(r0 + N);
            *(unsigned*)&Bs[n + 0][kk] = pack2(f0.x, f1.x);
            *(unsigned*)&Bs[n + 1][kk] = pack2(f0.y, f1.y);
            *(unsigned*)&Bs[n + 2][kk] = pack2(f0.z, f1.z);
            *(unsigned*)&Bs[n + 3][kk] = pack2(f0.w, f1.w);
        }
        // prefetch next K-tile into caches (global_prefetch_b8)
        if (k0 + 32 < K) {
            if (t < 128) {
                if (A_BF16)
                    __builtin_prefetch((const unsigned short*)Aptr + (size_t)(bm + t) * K + k0 + 32, 0, 0);
                else
                    __builtin_prefetch((const float*)Aptr + (size_t)(bm + t) * K + k0 + 32, 0, 0);
            } else if (t < 160) {
                __builtin_prefetch(Bw + (size_t)(k0 + 32 + (t - 128)) * N + bn, 0, 0);
            }
        }
        __syncthreads();

        FragU fa;
        const unsigned short* ar = &As[wave * 16 + li][0];
        fa.q[0] = *(const vu4*)(ar + hl * 8);
        fa.q[1] = *(const vu4*)(ar + 16 + hl * 8);
        for (int nt = 0; nt < 4; ++nt) {
            FragU fb;
            const unsigned short* br = &Bs[nt * 16 + li][0];
            fb.q[0] = *(const vu4*)(br + hl * 8);
            fb.q[1] = *(const vu4*)(br + 16 + hl * 8);
            acc[nt] = __builtin_amdgcn_wmma_f32_16x16x32_bf16(
                false, fa.v, false, fb.v, (short)0, acc[nt], false, false);
        }
        __syncthreads();
    }

    // epilogue
    for (int nt = 0; nt < 4; ++nt) {
        int col = bn + nt * 16 + li;
        for (int r = 0; r < 8; ++r) {
            int row = bm + wave * 16 + r + hl * 8;
            float v = acc[nt][r] + bias[col];
            if (HAS_RES) v += res[(size_t)row * N + col];
            if (RELU) v = fmaxf(v, 0.f);
            if (OUT_BF16)
                ((unsigned short*)Out)[(size_t)row * N + col] = f2bf(v);
            else
                ((float*)Out)[(size_t)row * N + col] = v;
        }
    }
}

// ---------------- LayerNorm (wave per row, b128 I/O) ----------------
__global__ __launch_bounds__(256) void ln_kernel(
    const float* __restrict__ in, const float* __restrict__ g,
    const float* __restrict__ b, float* __restrict__ out, int width)
{
    int row = blockIdx.x * 8 + (threadIdx.x >> 5);
    int lane = threadIdx.x & 31;
    int per = width >> 5;                       // 16 (H) or 8 (D)
    int nv = per >> 2;                          // float4s per lane
    const float* p = in + (size_t)row * width + lane * per;
    vf4 vals[4];
    float s = 0.f;
    for (int i = 0; i < nv; ++i) {
        vals[i] = *(const vf4*)(p + i * 4);
        s += vals[i].x + vals[i].y + vals[i].z + vals[i].w;
    }
    for (int off = 16; off > 0; off >>= 1) s += __shfl_xor(s, off);
    float mean = s / (float)width;
    float vs = 0.f;
    for (int i = 0; i < nv; ++i) {
        vf4 d = vals[i] - mean;
        vs += d.x * d.x + d.y * d.y + d.z * d.z + d.w * d.w;
    }
    for (int off = 16; off > 0; off >>= 1) vs += __shfl_xor(vs, off);
    float rstd = rsqrtf(vs / (float)width + 1e-5f);
    float* q = out + (size_t)row * width + lane * per;
    for (int i = 0; i < nv; ++i) {
        vf4 gg = *(const vf4*)(g + lane * per + i * 4);
        vf4 bb = *(const vf4*)(b + lane * per + i * 4);
        vf4 o = (vals[i] - mean) * rstd * gg + bb;
        *(vf4*)(q + i * 4) = o;
    }
}

// ---------------- Social pooling finalize: x += 0.2*soc ----------------
__global__ __launch_bounds__(256) void soc_finalize_kernel(
    const float* __restrict__ pre, const float* __restrict__ g,
    const float* __restrict__ b, const int* __restrict__ ncnt,
    const int* __restrict__ src, float* __restrict__ x)
{
    int row = blockIdx.x * 8 + (threadIdx.x >> 5);
    int lane = threadIdx.x & 31;
    const float* p = pre + (size_t)row * DD + lane * 8;
    vf4 vals[2];
    float s = 0.f;
    for (int i = 0; i < 2; ++i) {
        vals[i] = *(const vf4*)(p + i * 4);
        s += vals[i].x + vals[i].y + vals[i].z + vals[i].w;
    }
    for (int off = 16; off > 0; off >>= 1) s += __shfl_xor(s, off);
    float mean = s / (float)DD;
    float vs = 0.f;
    for (int i = 0; i < 2; ++i) {
        vf4 d = vals[i] - mean;
        vs += d.x * d.x + d.y * d.y + d.z * d.z + d.w * d.w;
    }
    for (int off = 16; off > 0; off >>= 1) vs += __shfl_xor(vs, off);
    float rstd = rsqrtf(vs / (float)DD + 1e-5f);
    int cnt = ncnt[row];
    int tok = src[row];
    float* xr = x + (size_t)row * DD + lane * 8;
    for (int i = 0; i < 2; ++i) {
        vf4 gg = *(const vf4*)(g + lane * 8 + i * 4);
        vf4 bb = *(const vf4*)(b + lane * 8 + i * 4);
        vf4 flat = *(const vf4*)(xr + i * 4);
        vf4 lnv = (vals[i] - mean) * rstd * gg + bb;
        vf4 soc;
        for (int e = 0; e < 4; ++e) {
            float lv = fmaxf(lnv[e], 0.f);
            float sv = (cnt > 0) ? lv : flat[e];
            soc[e] = (tok == PAD) ? 0.f : sv;
        }
        *(vf4*)(xr + i * 4) = flat + 0.2f * soc;
    }
}

// ---------------- Fused attention: one wave per (b, head, 16-query tile) ----------------
__global__ __launch_bounds__(32) void attn_kernel(
    const float* __restrict__ q, const float* __restrict__ k,
    const float* __restrict__ v, const int* __restrict__ src,
    float* __restrict__ o)
{
    __shared__ float sc[16][SS];                  // 32 KB scores
    __shared__ unsigned short pb[16][SS];         // 16 KB bf16 probs
    __shared__ unsigned short ks[16][LSTR];       // K tile (16 keys x 64 hd)
    __shared__ unsigned short vt[64][LSTR];       // transposed V tile (hd x 32 keys)
    __shared__ float madd[SS];                    // additive key mask (0 / -1e10)

    int blk = blockIdx.x;
    int qt = blk & 31;
    int head = (blk >> 5) & 7;
    int b = blk >> 8;
    int lane = threadIdx.x;
    int hl = lane >> 4, li = lane & 15;
    int bS = b * SS;
    int qrow0 = bS + qt * 16;

    // one-time: stage additive mask for all 512 keys
    for (int i = 0; i < 16; ++i) {
        int key = lane + i * 32;
        madd[key] = (src[bS + key] != PAD) ? 0.f : -1e10f;
    }

    // ---- Q fragments (A role): 16 rows x 64, two K-halves of 32 ----
    FragU qa[2];
    {
        const float* qp = q + (size_t)(qrow0 + li) * HH + head * HDD;
        for (int kh = 0; kh < 2; ++kh) {
            int kk0 = kh * 32 + hl * 8;
            vf4 f0 = *(const vf4*)(qp + kk0);
            vf4 f1 = *(const vf4*)(qp + kk0 + 4);
            vf4 f2 = *(const vf4*)(qp + kk0 + 16);
            vf4 f3 = *(const vf4*)(qp + kk0 + 20);
            qa[kh].u[0] = pack2(f0.x, f0.y); qa[kh].u[1] = pack2(f0.z, f0.w);
            qa[kh].u[2] = pack2(f1.x, f1.y); qa[kh].u[3] = pack2(f1.z, f1.w);
            qa[kh].u[4] = pack2(f2.x, f2.y); qa[kh].u[5] = pack2(f2.z, f2.w);
            qa[kh].u[6] = pack2(f3.x, f3.y); qa[kh].u[7] = pack2(f3.z, f3.w);
        }
    }

    // ---- scores: Q @ K^T over 32 key tiles of 16 (K staged in LDS) ----
    for (int kt = 0; kt < 32; ++kt) {
        {   // cooperative coalesced stage: 2 lanes per key row
            int key = lane >> 1, half = lane & 1;
            const float* kp = k + (size_t)(bS + kt * 16 + key) * HH + head * HDD + half * 32;
            for (int j = 0; j < 8; ++j) {
                vf4 f = *(const vf4*)(kp + j * 4);
                int kk = half * 32 + j * 4;
                *(unsigned*)&ks[key][kk]     = pack2(f.x, f.y);
                *(unsigned*)&ks[key][kk + 2] = pack2(f.z, f.w);
            }
        }
        __syncthreads();
        f32x8 c;
        for (int r = 0; r < 8; ++r) c[r] = 0.f;
        for (int kh = 0; kh < 2; ++kh) {
            FragU fb;
            const unsigned short* br = &ks[li][kh * 32 + hl * 8];
            fb.q[0] = *(const vu4*)br;
            fb.q[1] = *(const vu4*)(br + 16);
            c = __builtin_amdgcn_wmma_f32_16x16x32_bf16(
                false, qa[kh].v, false, fb.v, (short)0, c, false, false);
        }
        int key = kt * 16 + li;
        float ma = madd[key];
        for (int r = 0; r < 8; ++r)
            sc[r + hl * 8][key] = c[r] * 0.125f + ma;
        __syncthreads();
    }

    // ---- softmax: 2 half-lanes per row, 256 cols each, b128 LDS I/O ----
    {
        int row = li, c0 = hl * 256;
        vf4* scr = (vf4*)&sc[row][c0];
        float mx = -3.4e38f;
        for (int i = 0; i < 64; ++i) {
            vf4 tv = scr[i];
            mx = fmaxf(mx, fmaxf(fmaxf(tv.x, tv.y), fmaxf(tv.z, tv.w)));
        }
        mx = fmaxf(mx, __shfl_xor(mx, 16));
        float sum = 0.f;
        for (int i = 0; i < 64; ++i) {
            vf4 tv = scr[i];
            tv.x = __expf(tv.x - mx); tv.y = __expf(tv.y - mx);
            tv.z = __expf(tv.z - mx); tv.w = __expf(tv.w - mx);
            scr[i] = tv;
            sum += tv.x + tv.y + tv.z + tv.w;
        }
        sum += __shfl_xor(sum, 16);
        float inv = 1.f / sum;
        vu2* pbr = (vu2*)&pb[row][c0];
        for (int i = 0; i < 64; ++i) {
            vf4 tv = scr[i] * inv;
            vu2 w;
            w.x = pack2(tv.x, tv.y);
            w.y = pack2(tv.z, tv.w);
            pbr[i] = w;
        }
    }
    __syncthreads();

    // ---- P @ V: 16 k-steps of 32 keys, V staged transposed (key pairs packed) ----
    f32x8 acc[4];
    for (int nt = 0; nt < 4; ++nt)
        for (int r = 0; r < 8; ++r) acc[nt][r] = 0.f;

    for (int kt = 0; kt < 16; ++kt) {
        {   // lane handles 2 consecutive keys x 32 hd
            int kk2 = (lane & 15) * 2, hh = (lane >> 4) * 32;
            const float* vp0 = v + (size_t)(bS + kt * 32 + kk2) * HH + head * HDD + hh;
            const float* vp1 = vp0 + HH;
            for (int j = 0; j < 8; ++j) {
                vf4 a = *(const vf4*)(vp0 + j * 4);
                vf4 bb = *(const vf4*)(vp1 + j * 4);
                int hd = hh + j * 4;
                *(unsigned*)&vt[hd + 0][kk2] = pack2(a.x, bb.x);
                *(unsigned*)&vt[hd + 1][kk2] = pack2(a.y, bb.y);
                *(unsigned*)&vt[hd + 2][kk2] = pack2(a.z, bb.z);
                *(unsigned*)&vt[hd + 3][kk2] = pack2(a.w, bb.w);
            }
        }
        __syncthreads();

        FragU fa;
        const unsigned short* pr = &pb[li][kt * 32];
        fa.q[0] = *(const vu4*)(pr + hl * 8);
        fa.q[1] = *(const vu4*)(pr + 16 + hl * 8);
        for (int nt = 0; nt < 4; ++nt) {
            FragU fb;
            const unsigned short* br = &vt[nt * 16 + li][hl * 8];
            fb.q[0] = *(const vu4*)br;
            fb.q[1] = *(const vu4*)(br + 16);
            acc[nt] = __builtin_amdgcn_wmma_f32_16x16x32_bf16(
                false, fa.v, false, fb.v, (short)0, acc[nt], false, false);
        }
        __syncthreads();
    }

    // ---- write O tile ----
    for (int nt = 0; nt < 4; ++nt) {
        int col = head * HDD + nt * 16 + li;
        for (int r = 0; r < 8; ++r) {
            int row = qrow0 + r + hl * 8;
            o[(size_t)row * HH + col] = acc[nt][r];
        }
    }
}

// ---------------- Host orchestration ----------------
extern "C" void kernel_launch(void* const* d_in, const int* in_sizes, int n_in,
                              void* d_out, int out_size, void* d_ws, size_t ws_size,
                              hipStream_t stream) {
    (void)in_sizes; (void)n_in; (void)out_size; (void)ws_size;
    const int*   src   = (const int*)d_in[0];
    const int*   nidx  = (const int*)d_in[2];
    const int*   ncnt  = (const int*)d_in[3];
    const float* tint  = (const float*)d_in[4];
    const float* emb   = (const float*)d_in[5];
    const float* timeW = (const float*)d_in[6];
    const float* timeB = (const float*)d_in[7];
    const float* socW  = (const float*)d_in[8];
    const float* socB  = (const float*)d_in[9];
    const float* socG  = (const float*)d_in[10];
    const float* socBt = (const float*)d_in[11];
    const float* projW = (const float*)d_in[12];
    const float* projB = (const float*)d_in[13];
    const float* Wq = (const float*)d_in[14]; const float* bq = (const float*)d_in[15];
    const float* Wk = (const float*)d_in[16]; const float* bk = (const float*)d_in[17];
    const float* Wv = (const float*)d_in[18]; const float* bv = (const float*)d_in[19];
    const float* Wo = (const float*)d_in[20]; const float* bo = (const float*)d_in[21];
    const float* ln1g = (const float*)d_in[22]; const float* ln1b = (const float*)d_in[23];
    const float* ffW1 = (const float*)d_in[24]; const float* ffb1 = (const float*)d_in[25];
    const float* ffW2 = (const float*)d_in[26]; const float* ffb2 = (const float*)d_in[27];
    const float* ln2g = (const float*)d_in[28]; const float* ln2b = (const float*)d_in[29];

    // outputs: h [N,H] then x [N,D]
    float* hbuf = (float*)d_out;
    float* xbuf = hbuf + (size_t)NTOK * HH;

    // workspace layout (floats): q,k,v,o,t1 (N*H each), then ff1 bf16 (N*PF)
    const size_t NH_ = (size_t)NTOK * HH;
    float* qbuf = (float*)d_ws;
    float* kbuf = qbuf + NH_;
    float* vbuf = kbuf + NH_;
    float* obuf = vbuf + NH_;
    float* t1   = obuf + NH_;
    float* agg  = t1;
    float* socp = t1 + (size_t)NTOK * DD;
    unsigned short* ffb = (unsigned short*)(t1 + NH_);

    embed_kernel<<<NTOK, DD, 0, stream>>>(src, tint, emb, timeW, timeB, xbuf);
    agg_kernel<<<NTOK, DD, 0, stream>>>(xbuf, nidx, ncnt, agg);
    gemm_kernel<false, false, false, true><<<dim3(DD / 64, NTOK / 128), 256, 0, stream>>>(
        agg, socW, socB, xbuf, socp, NTOK, DD, DD);
    soc_finalize_kernel<<<NTOK / 8, 256, 0, stream>>>(socp, socG, socBt, ncnt, src, xbuf);
    gemm_kernel<false, false, false, false><<<dim3(HH / 64, NTOK / 128), 256, 0, stream>>>(
        xbuf, projW, projB, nullptr, hbuf, NTOK, HH, DD);

    for (int l = 0; l < LL; ++l) {
        const float* Wq_l = Wq + (size_t)l * HH * HH;
        const float* Wk_l = Wk + (size_t)l * HH * HH;
        const float* Wv_l = Wv + (size_t)l * HH * HH;
        const float* Wo_l = Wo + (size_t)l * HH * HH;
        const float* W1_l = ffW1 + (size_t)l * HH * PFF;
        const float* W2_l = ffW2 + (size_t)l * PFF * HH;

        dim3 gHH(HH / 64, NTOK / 128);
        gemm_kernel<false, false, false, false><<<gHH, 256, 0, stream>>>(
            hbuf, Wq_l, bq + l * HH, nullptr, qbuf, NTOK, HH, HH);
        gemm_kernel<false, false, false, false><<<gHH, 256, 0, stream>>>(
            hbuf, Wk_l, bk + l * HH, nullptr, kbuf, NTOK, HH, HH);
        gemm_kernel<false, false, false, false><<<gHH, 256, 0, stream>>>(
            hbuf, Wv_l, bv + l * HH, nullptr, vbuf, NTOK, HH, HH);

        attn_kernel<<<BB * NHH * (SS / 16), 32, 0, stream>>>(qbuf, kbuf, vbuf, src, obuf);

        gemm_kernel<false, false, false, true><<<gHH, 256, 0, stream>>>(
            obuf, Wo_l, bo + l * HH, hbuf, t1, NTOK, HH, HH);
        ln_kernel<<<NTOK / 8, 256, 0, stream>>>(t1, ln1g + l * HH, ln1b + l * HH, hbuf, HH);

        gemm_kernel<false, true, true, false><<<dim3(PFF / 64, NTOK / 128), 256, 0, stream>>>(
            hbuf, W1_l, ffb1 + l * PFF, nullptr, ffb, NTOK, PFF, HH);
        gemm_kernel<true, false, false, true><<<gHH, 256, 0, stream>>>(
            ffb, W2_l, ffb2 + l * HH, hbuf, t1, NTOK, HH, PFF);
        ln_kernel<<<NTOK / 8, 256, 0, stream>>>(t1, ln2g + l * HH, ln2b + l * HH, hbuf, HH);
    }
}
